// LocalDecoder_19816979104332
// MI455X (gfx1250) — matrix operand order repaired and checked by reading, not compiled, not run on hardware
//
#include <hip/hip_runtime.h>

// ---------------------------------------------------------------------------
// MI455X (gfx1250) fused LocalDecoder:
//   trilinear grid_sample (fp32 gather)  +  5x ResnetBlockFC MLP via
//   v_wmma_f32_16x16x32_bf16, weights staged to LDS with async-to-LDS loads.
// Biases for fresh-accumulator GEMMs are folded into a constant-1 K-column
// so the WMMA stream carries no bias splat/mov overhead.
// ---------------------------------------------------------------------------

#define ASYNC_STAGE 1   // verified: assembler accepts the async mnemonic

typedef unsigned short u16;
typedef unsigned int   u32;
typedef unsigned long long u64;
typedef __attribute__((ext_vector_type(16))) __bf16 v16bf;
typedef __attribute__((ext_vector_type(8)))  __bf16 v8bf;
typedef __attribute__((ext_vector_type(8)))  float  v8f;
typedef __attribute__((ext_vector_type(8)))  u16    v8us;

#define BF16_ONE ((u16)0x3F80)

// ---- workspace layout (bf16 elements) -------------------------------------
// B-fragment swizzled weights. Fragment f = kt*16 + nt, 512 elems each:
//   elem(lane,j):  n = nt*16 + (lane&15),  k = kt*32 + (lane>>4)*16 + j
#define WP_OFF 0               // fc_p_w+fc_p_b : 16 nt x 1 kt -> 8192  (k==3 is bias row)
#define WC_OFF 8192            // fc_c_w        : 5 x (16x4x512) = 163840
#define B0_OFF 172032          // b0_w+b0_b     : 5 x (16x9x512) = 368640 (k==256 bias row)
#define B1_OFF 540672          // b1_w          : 5 x (16x8x512) = 327680
#define TOTW   868352          // 1.74 MB of workspace as bf16

// LDS partition (u16 elements): shared weight buffer + per-wave activations
#define WLDS_ELEMS 73728       // 144 KB, max layer (b0: 288x256 bf16)
#define ACT_P 512              // 16 x 32   (K padded; col 3 = 1.0 bias col)
#define ACT_C 2048             // 16 x 128
#define ACT_H 4608             // 16 x 288  (col 256 = 1.0 bias col)
#define ACT_G 4096             // 16 x 256
#define ACT_PER_WAVE (ACT_P + ACT_C + ACT_H + ACT_G)       // 11264
#define SMEM_ELEMS (WLDS_ELEMS + 8 * ACT_PER_WAVE)         // 163840
#define SMEM_BYTES (SMEM_ELEMS * 2)                        // 327680 == 320 KB

__device__ __forceinline__ u16 f2bf(float f) {           // fp32 -> bf16 RNE
  u32 u = __float_as_uint(f);
  u += 0x7fffu + ((u >> 16) & 1u);
  return (u16)(u >> 16);
}

__device__ __forceinline__ v8f wmma_bf16(v16bf a, v16bf b, v8f c) {
  return __builtin_amdgcn_wmma_f32_16x16x32_bf16(false, a, false, b,
                                                 (short)0, c, false, false);
}

// A fragment (16x32 bf16, M over lanes 0..15) from row-major LDS [m][K]:
//  lane<16 : elems0-15 = K..K+15
//  lane>=16: elems0-15 = K+16..K+31
__device__ __forceinline__ v16bf loadA(const u16* act, int K, int kt, int lane) {
  const u16* ptr = act + (lane & 15) * K + kt * 32 + (lane >> 4) * 16;
  v8bf lo = *(const v8bf*)(ptr);
  v8bf hi = *(const v8bf*)(ptr + 8);
  return __builtin_shufflevector(lo, hi, 0,1,2,3,4,5,6,7,8,9,10,11,12,13,14,15);
}

// B fragment: contiguous 32 bytes per lane from pre-swizzled LDS weights
__device__ __forceinline__ v16bf loadB(const u16* w, int kt, int nt, int lane) {
  return *(const v16bf*)(w + ((kt * 16 + nt) << 9) + lane * 16);
}

__device__ __forceinline__ void add_bias(v8f& c, const float* b, int off, int t, int lane) {
  float bv = b[off + t * 16 + (lane & 15)];
#pragma unroll
  for (int r = 0; r < 8; ++r) c[r] += bv;
}

// relu(C-frag) -> bf16 row-major LDS [m][stride]  (C layout: M=r+8*(lane>>4), N=t*16+(lane&15))
__device__ __forceinline__ void store_relu_tile(u16* dst, int stride, int t,
                                                v8f acc, int lane) {
  int n  = t * 16 + (lane & 15);
  int mb = (lane >> 4) * 8;
#pragma unroll
  for (int r = 0; r < 8; ++r) {
    float v = acc[r] > 0.f ? acc[r] : 0.f;
    dst[(mb + r) * stride + n] = f2bf(v);
  }
}

// cooperative: stage `count` bf16 elems of weights into shared LDS buffer
__device__ __forceinline__ void stage_weights(const u16* __restrict__ src,
                                              u16* dst, int count) {
#if ASYNC_STAGE
  for (int o = threadIdx.x * 8; o < count; o += 2048) {
    u32 lds = (u32)(u64)(const void*)(dst + o);        // low 32b = LDS offset
    u64 ga  = (u64)(const void*)(src + o);
    asm volatile("global_load_async_to_lds_b128 %0, %1, off"
                 :: "v"(lds), "v"(ga) : "memory");
  }
  asm volatile("s_wait_asynccnt 0x0" ::: "memory");
#else
  for (int o = threadIdx.x * 8; o < count; o += 2048)
    *(v8us*)(dst + o) = *(const v8us*)(src + o);
#endif
}

// ---------------------------------------------------------------------------
// Kernel 1: fp32 weights (+fused biases) -> bf16 B-fragment layout in ws
// ---------------------------------------------------------------------------
__device__ __forceinline__ void fragNK(int rel, int& n, int& k) {
  int f = rel >> 9, e = rel & 511;
  int kt = f >> 4, nt = f & 15;
  int lane = e >> 4, j = e & 15;
  n = nt * 16 + (lane & 15);
  k = kt * 32 + ((lane >> 4) << 4) + j;
}

__global__ __launch_bounds__(256) void convert_weights(
    const float* __restrict__ wp, const float* __restrict__ bp,
    const float* __restrict__ wc,
    const float* __restrict__ w0, const float* __restrict__ b0,
    const float* __restrict__ w1,
    u16* __restrict__ dst) {
  int i = blockIdx.x * 256 + threadIdx.x;
  if (i >= TOTW) return;
  float v;
  int n, k;
  if (i < WC_OFF) {                         // fc_p_w (256x3, K pad 32, k==3 bias)
    fragNK(i, n, k);
    v = (k < 3) ? wp[n * 3 + k] : (k == 3 ? bp[n] : 0.f);
  } else if (i < B0_OFF) {                  // fc_c_w (5,256,128)
    int rel = i - WC_OFF;
    int L = rel / 32768, r2 = rel % 32768;
    fragNK(r2, n, k);
    v = wc[L * 32768 + n * 128 + k];
  } else if (i < B1_OFF) {                  // b0_w (5,256,256) + bias row k==256
    int rel = i - B0_OFF;
    int L = rel / 73728, r2 = rel % 73728;
    fragNK(r2, n, k);
    v = (k < 256) ? w0[L * 65536 + n * 256 + k]
                  : (k == 256 ? b0[L * 256 + n] : 0.f);
  } else {                                  // b1_w (5,256,256)
    int rel = i - B1_OFF;
    int L = rel >> 16, r2 = rel & 65535;
    fragNK(r2, n, k);
    v = w1[L * 65536 + n * 256 + k];
  }
  dst[i] = f2bf(v);
}

// ---------------------------------------------------------------------------
// Kernel 2: fused sample + MLP. 256 thr = 8 wave32, 16 points per wave.
// ---------------------------------------------------------------------------
__global__ __launch_bounds__(256) void decoder_kernel(
    const float* __restrict__ p, const float* __restrict__ cg,
    const float* __restrict__ fccb, const float* __restrict__ b1b,
    const float* __restrict__ outw, const float* __restrict__ outb,
    const u16* __restrict__ wbf, float* __restrict__ out,
    int npts, int Nn) {
  extern __shared__ u16 smem[];
  u16* wlds = smem;
  const int lane = threadIdx.x & 31;
  const int wave = threadIdx.x >> 5;
  u16* actP = smem + WLDS_ELEMS + wave * ACT_PER_WAVE;
  u16* actC = actP + ACT_P;
  u16* actH = actC + ACT_C;
  u16* actG = actH + ACT_H;

  const int m  = lane & 15;      // point row within wave tile
  const int hh = lane >> 4;      // channel half / M-half selector
  const int Pbase = blockIdx.x * 128 + wave * 16;
  int P  = Pbase + m;
  int Pl = P < npts ? P : npts - 1;

  // ---- trilinear sample setup (normalize_3d_coordinate + border clamp) ----
  const float inv = 1.0f / 1.101f;                 // 1/(1+PADDING+1e-3)
  float px = p[3 * Pl], py = p[3 * Pl + 1], pz = p[3 * Pl + 2];
  float fx = fminf(fmaxf((px * inv + 0.5f) * 63.f, 0.f), 63.f);
  float fy = fminf(fmaxf((py * inv + 0.5f) * 63.f, 0.f), 63.f);
  float fz = fminf(fmaxf((pz * inv + 0.5f) * 63.f, 0.f), 63.f);
  int x0 = (int)floorf(fx), y0 = (int)floorf(fy), z0 = (int)floorf(fz);
  float wx = fx - x0, wy = fy - y0, wz = fz - z0;
  int x1 = min(x0 + 1, 63), y1 = min(y0 + 1, 63), z1 = min(z0 + 1, 63);
  int s000 = (z0 * 64 + y0) * 64 + x0, s001 = (z0 * 64 + y0) * 64 + x1;
  int s010 = (z0 * 64 + y1) * 64 + x0, s011 = (z0 * 64 + y1) * 64 + x1;
  int s100 = (z1 * 64 + y0) * 64 + x0, s101 = (z1 * 64 + y0) * 64 + x1;
  int s110 = (z1 * 64 + y1) * 64 + x0, s111 = (z1 * 64 + y1) * 64 + x1;
  float w000 = (1 - wx) * (1 - wy) * (1 - wz), w001 = wx * (1 - wy) * (1 - wz);
  float w010 = (1 - wx) * wy * (1 - wz),       w011 = wx * wy * (1 - wz);
  float w100 = (1 - wx) * (1 - wy) * wz,       w101 = wx * (1 - wy) * wz;
  float w110 = (1 - wx) * wy * wz,             w111 = wx * wy * wz;
  int bi = Pl / Nn;

  // p-tile (16x32): k=0..2 = p, k=3 = 1.0 (bias column), rest 0
  v8us zz = (v8us)0;
  *(v8us*)(actP + lane * 16) = zz;
  if (lane < 16) {
    actP[m * 32 + 0] = f2bf(px);
    actP[m * 32 + 1] = f2bf(py);
    actP[m * 32 + 2] = f2bf(pz);
    actP[m * 32 + 3] = BF16_ONE;
  }

  // actH bias K-tile (cols 256..287): zero pad, col 256 = 1.0 (set once)
  if (lane < 16) {
#pragma unroll
    for (int q = 0; q < 4; ++q)
      *(v8us*)(actH + m * 288 + 256 + q * 8) = zz;
    actH[m * 288 + 256] = BF16_ONE;
  }

  // c-tile (16x128): lane pair per point, 64 channels each, 8 corners/fma
  for (int c0 = 0; c0 < 64; c0 += 8) {
    v8us pv;
#pragma unroll
    for (int c = 0; c < 8; ++c) {
      const float* g = cg + (size_t)(bi * 128 + hh * 64 + c0 + c) * 262144;
      float v = w000 * g[s000] + w001 * g[s001] + w010 * g[s010] + w011 * g[s011]
              + w100 * g[s100] + w101 * g[s101] + w110 * g[s110] + w111 * g[s111];
      pv[c] = f2bf(v);
    }
    *(v8us*)(actC + m * 128 + hh * 64 + c0) = pv;
  }

  // ---- net = [p,1] @ [fc_p_w;fc_p_b]^T  (one K-tile, 16 N-tiles) ----
  __syncthreads();
  stage_weights(wbf + WP_OFF, wlds, 8192);
  __syncthreads();

  v8f net[16];
  {
    v16bf a = loadA(actP, 32, 0, lane);
    v8f z = (v8f)0.0f;
#pragma unroll
    for (int t = 0; t < 16; ++t)
      net[t] = wmma_bf16(a, loadB(wlds, 0, t, lane), z);
  }

  // ---- 5 ResNet blocks ----
#pragma unroll 1
  for (int blk = 0; blk < 5; ++blk) {
    // net += c @ fc_c_w[blk]^T + fc_c_b[blk]
    __syncthreads();
    stage_weights(wbf + WC_OFF + blk * 32768, wlds, 32768);
    __syncthreads();
#pragma unroll 1
    for (int kt = 0; kt < 4; ++kt) {
      v16bf a = loadA(actC, 128, kt, lane);
#pragma unroll
      for (int t = 0; t < 16; ++t)
        net[t] = wmma_bf16(a, loadB(wlds, kt, t, lane), net[t]);
    }
#pragma unroll
    for (int t = 0; t < 16; ++t) add_bias(net[t], fccb, blk * 256, t, lane);

    // relu(net) -> actH (stride 288; bias column 256 stays 1.0)
#pragma unroll
    for (int t = 0; t < 16; ++t) store_relu_tile(actH, 288, t, net[t], lane);

    // h = [relu(net),1] @ [b0_w;b0_b]^T ; relu(h) -> actG  (9 K-tiles)
    __syncthreads();
    stage_weights(wbf + B0_OFF + blk * 73728, wlds, 73728);
    __syncthreads();
#pragma unroll 1
    for (int tc = 0; tc < 4; ++tc) {
      v8f acc[4];
      v8f z = (v8f)0.0f;
#pragma unroll
      for (int ti = 0; ti < 4; ++ti) acc[ti] = z;
#pragma unroll 1
      for (int kt = 0; kt < 9; ++kt) {
        v16bf a = loadA(actH, 288, kt, lane);
#pragma unroll
        for (int ti = 0; ti < 4; ++ti)
          acc[ti] = wmma_bf16(a, loadB(wlds, kt, tc * 4 + ti, lane), acc[ti]);
      }
#pragma unroll
      for (int ti = 0; ti < 4; ++ti)
        store_relu_tile(actG, 256, tc * 4 + ti, acc[ti], lane);
    }

    // net += relu(h) @ b1_w^T + b1_b   (accumulate straight into net frags)
    __syncthreads();
    stage_weights(wbf + B1_OFF + blk * 65536, wlds, 65536);
    __syncthreads();
#pragma unroll 1
    for (int kt = 0; kt < 8; ++kt) {
      v16bf a = loadA(actG, 256, kt, lane);
#pragma unroll
      for (int t = 0; t < 16; ++t)
        net[t] = wmma_bf16(a, loadB(wlds, kt, t, lane), net[t]);
    }
#pragma unroll
    for (int t = 0; t < 16; ++t) add_bias(net[t], b1b, blk * 256, t, lane);
  }

  // ---- out = relu(net) @ out_w^T + out_b : per-lane partial + 16-lane xor ----
  float pr[8];
#pragma unroll
  for (int r = 0; r < 8; ++r) pr[r] = 0.f;
#pragma unroll
  for (int t = 0; t < 16; ++t) {
    float wv = outw[t * 16 + (lane & 15)];
#pragma unroll
    for (int r = 0; r < 8; ++r) {
      float v = net[t][r];
      pr[r] += (v > 0.f ? v : 0.f) * wv;
    }
  }
#pragma unroll
  for (int d = 1; d < 16; d <<= 1)
#pragma unroll
    for (int r = 0; r < 8; ++r) pr[r] += __shfl_xor(pr[r], d, 32);

  if ((lane & 15) == 0) {
    float ob = outb[0];
#pragma unroll
    for (int r = 0; r < 8; ++r) {
      int Po = Pbase + hh * 8 + r;
      if (Po < npts) out[Po] = pr[r] + ob;
    }
  }
}

// ---------------------------------------------------------------------------
extern "C" void kernel_launch(void* const* d_in, const int* in_sizes, int n_in,
                              void* d_out, int out_size, void* d_ws, size_t ws_size,
                              hipStream_t stream) {
  const float* p    = (const float*)d_in[0];
  const float* cg   = (const float*)d_in[1];
  const float* fcpw = (const float*)d_in[2];
  const float* fcpb = (const float*)d_in[3];
  const float* fccw = (const float*)d_in[4];
  const float* fccb = (const float*)d_in[5];
  const float* b0w  = (const float*)d_in[6];
  const float* b0b  = (const float*)d_in[7];
  const float* b1w  = (const float*)d_in[8];
  const float* b1b  = (const float*)d_in[9];
  const float* outw = (const float*)d_in[10];
  const float* outb = (const float*)d_in[11];
  float* out = (float*)d_out;
  u16* wbf = (u16*)d_ws;   // needs TOTW*2 = 1.74 MB of workspace

  int npts = in_sizes[0] / 3;                       // B*N
  int Bc = in_sizes[1] / (128 * 64 * 64 * 64);      // batch from grid size
  if (Bc < 1) Bc = 1;
  int Nn = npts / Bc;

  convert_weights<<<(TOTW + 255) / 256, 256, 0, stream>>>(
      fcpw, fcpb, fccw, b0w, b0b, b1w, wbf);

  int nblk = (npts + 127) / 128;
  decoder_kernel<<<nblk, 256, SMEM_BYTES, stream>>>(
      p, cg, fccb, b1b, outw, outb, wbf, out, npts, Nn);
}
